// SelfAttention_54382875902080
// MI455X (gfx1250) — compile-verified, matrix operand order checked by reading, not run on hardware
//
#include <hip/hip_runtime.h>
#include <hip/hip_bf16.h>

typedef __bf16 bf16_t;
typedef __attribute__((ext_vector_type(16))) __bf16 v16bf;
typedef __attribute__((ext_vector_type(8)))  float  v8f;

// D = A(16x32 bf16) x B(32x16 bf16) + C(16x16 f32)
__device__ __forceinline__ v8f wmma_bf16(v16bf a, v16bf b, v8f c) {
  return __builtin_amdgcn_wmma_f32_16x16x32_bf16(
      /*neg_a=*/false, a, /*neg_b=*/false, b,
      /*c_mod=*/(short)0, c, /*reuse_a=*/false, /*reuse_b=*/false);
}

// A/B fragment for 16-bit WMMA, built from an f32 row in global memory.
// rowbase points at element [row][k0] of a row-major matrix.
// Element j of the frag maps to k = (j<8) ? 8*half + j : 16 + 8*half + (j-8).
__device__ __forceinline__ v16bf frag_from_f32(const float* rowbase, int hlf) {
  v16bf f;
  const float* p0 = rowbase + 8 * hlf;
  const float* p1 = rowbase + 16 + 8 * hlf;
#pragma unroll
  for (int j = 0; j < 8; ++j) {
    f[j]     = (bf16_t)p0[j];
    f[j + 8] = (bf16_t)p1[j];
  }
  return f;
}

// Same mapping, reading bf16 directly from LDS (two 16B contiguous runs).
__device__ __forceinline__ v16bf frag_from_lds(const bf16_t* rowbase, int hlf) {
  v16bf f;
  const bf16_t* p0 = rowbase + 8 * hlf;
  const bf16_t* p1 = rowbase + 16 + 8 * hlf;
#pragma unroll
  for (int j = 0; j < 8; ++j) {
    f[j]     = p0[j];
    f[j + 8] = p1[j];
  }
  return f;
}

// B-fragment from an f32 tile in LDS where k runs down rows (row stride =
// `stride` floats) and the n-column is fixed (baked into base).
__device__ __forceinline__ v16bf frag_from_lds_f32_strided(const float* base,
                                                           int stride, int hlf) {
  v16bf f;
  const float* p0 = base + (8 * hlf) * stride;
  const float* p1 = base + (16 + 8 * hlf) * stride;
#pragma unroll
  for (int j = 0; j < 8; ++j) {
    f[j]     = (bf16_t)p0[j * stride];
    f[j + 8] = (bf16_t)p1[j * stride];
  }
  return f;
}

// Async-copy a 32-row x 256B (64 f32) tile from global (row stride ldg floats)
// into a per-wave LDS buffer (row stride 64 floats). 16 x b128 per lane-wave,
// tracked by ASYNCcnt.
__device__ __forceinline__ void async_copy_vtile(const float* gbase, int ldg,
                                                 float* lbase, int lane) {
#pragma unroll
  for (int i = 0; i < 16; ++i) {
    const int linear = i * 32 + lane;   // unit = 16B chunk
    const int row    = linear >> 4;     // 16 chunks per 256B row
    const int chunk  = linear & 15;
    const float* g = gbase + (size_t)row * ldg + chunk * 4;
    const uint32_t l =
        (uint32_t)(uintptr_t)(lbase + row * 64 + chunk * 4);  // LDS offset
    asm volatile("global_load_async_to_lds_b128 %0, %1, off"
                 :: "v"(l), "v"(g) : "memory");
  }
}

// C = A[M,K] * B[N,K]^T + bias, all f32 in memory, bf16 WMMA internally.
// Block: 256 threads (8 waves), 128x128 tile, K-step 32.
__global__ __launch_bounds__(256) void gemm_nt_bias(
    const float* __restrict__ A, const float* __restrict__ B,
    const float* __restrict__ bias, float* __restrict__ C,
    int M, int N, int K) {
  __shared__ bf16_t lA[128 * 32];
  __shared__ bf16_t lB[128 * 32];

  const int tid  = threadIdx.x;
  const int lane = tid & 31;
  const int hlf  = lane >> 4;   // 0/1: lane half
  const int lm   = lane & 15;
  const int wid  = tid >> 5;

  const int bm = blockIdx.y * 128;
  const int bn = blockIdx.x * 128;
  const int wm = (wid & 3) * 32;   // wave M offset within tile
  const int wn = (wid >> 2) * 64;  // wave N offset within tile

  v8f acc[2][4];
#pragma unroll
  for (int mt = 0; mt < 2; ++mt)
#pragma unroll
    for (int nt = 0; nt < 4; ++nt)
#pragma unroll
      for (int r = 0; r < 8; ++r) acc[mt][nt][r] = 0.0f;

  const int lrow  = tid >> 1;   // 0..127
  const int lhalf = tid & 1;    // 0/1

  for (int k0 = 0; k0 < K; k0 += 32) {
    const float* ga = A + (size_t)(bm + lrow) * K + k0 + lhalf * 16;
    const float* gb = B + (size_t)(bn + lrow) * K + k0 + lhalf * 16;
    bf16_t* sa = lA + lrow * 32 + lhalf * 16;
    bf16_t* sb = lB + lrow * 32 + lhalf * 16;
#pragma unroll
    for (int j = 0; j < 16; ++j) {
      sa[j] = (bf16_t)ga[j];
      sb[j] = (bf16_t)gb[j];
    }
    if (k0 + 32 < K) {  // pull next K-slice toward the caches
      __builtin_prefetch(ga + 32, 0, 0);
      __builtin_prefetch(gb + 32, 0, 0);
    }
    __syncthreads();

    v16bf af[2], bfr[4];
#pragma unroll
    for (int mt = 0; mt < 2; ++mt)
      af[mt] = frag_from_lds(lA + (wm + mt * 16 + lm) * 32, hlf);
#pragma unroll
    for (int nt = 0; nt < 4; ++nt)
      bfr[nt] = frag_from_lds(lB + (wn + nt * 16 + lm) * 32, hlf);
#pragma unroll
    for (int mt = 0; mt < 2; ++mt)
#pragma unroll
      for (int nt = 0; nt < 4; ++nt)
        acc[mt][nt] = wmma_bf16(af[mt], bfr[nt], acc[mt][nt]);

    __syncthreads();
  }

  // epilogue: C frag layout is M = r + 8*half, N = lane%16
#pragma unroll
  for (int mt = 0; mt < 2; ++mt) {
#pragma unroll
    for (int nt = 0; nt < 4; ++nt) {
      const int gn = bn + wn + nt * 16 + lm;
      const float bv = bias[gn];
#pragma unroll
      for (int r = 0; r < 8; ++r) {
        const int gm = bm + wm + mt * 16 + r + 8 * hlf;
        C[(size_t)gm * N + gn] = acc[mt][nt][r] + bv;
      }
    }
  }
}

// Flash-style attention: one wave per (batch, head, 16-query tile).
// qkv layout: [B*S, 3072] with q at col h*64, k at 1024+h*64, v at 2048+h*64.
// V tiles are staged global->LDS with async b128 copies, double-buffered so
// the copy of key-block kb+32 overlaps the QK^T/softmax of block kb.
// Dynamic LDS: 8 waves * 2 * 8KB V buffers + 8 * 1KB P staging = 136KB.
__global__ __launch_bounds__(256) void attn_kernel(
    const float* __restrict__ qkv, float* __restrict__ score,
    const int* __restrict__ apply_mask_p) {
  constexpr int S = 2048, DH = 64, DEMB = 1024, LDQ = 3072;

  extern __shared__ char smem_raw[];

  const int tid  = threadIdx.x;
  const int lane = tid & 31;
  const int hlf  = lane >> 4;
  const int lm   = lane & 15;
  const int wid  = tid >> 5;

  float*  vbuf = (float*)smem_raw + (size_t)wid * 2 * 2048;     // 2 x 8KB
  bf16_t* pbuf = (bf16_t*)(smem_raw + 131072) + wid * 512;      // 1KB

  const int tile = blockIdx.x * 8 + wid;  // 0..4095
  const int qt = tile & 127;              // S/16 tiles
  const int h  = (tile >> 7) & 15;
  const int b  = tile >> 11;
  const int q0 = qt * 16;

  const int mask = *apply_mask_p;

  const float* qbase = qkv + (size_t)(b * S) * LDQ + h * DH;
  const float* kbase = qbase + DEMB;
  const float* vbase = qbase + 2 * DEMB;

  // Q A-frags (16 x 64 -> two 16x32 k-slices), loaded once
  v16bf qf[2];
#pragma unroll
  for (int ks = 0; ks < 2; ++ks)
    qf[ks] = frag_from_f32(qbase + (size_t)(q0 + lm) * LDQ + ks * 32, hlf);

  v8f o[4];
  float mrow[8], lrow[8];
#pragma unroll
  for (int r = 0; r < 8; ++r) { mrow[r] = -1e30f; lrow[r] = 0.0f; }
#pragma unroll
  for (int nt = 0; nt < 4; ++nt)
#pragma unroll
    for (int r = 0; r < 8; ++r) o[nt][r] = 0.0f;

  const int kend = mask ? (q0 + 16) : S;  // exclusive key bound (causal)

  // prologue: stage V tile for kb = 0 into buffer 0
  async_copy_vtile(vbase, LDQ, vbuf, lane);

  int ibuf = 0;
  for (int kb = 0; kb < kend; kb += 32, ibuf ^= 1) {
    float* vcur = vbuf + ibuf * 2048;
    const bool have_next = (kb + 32 < kend);
    if (have_next)  // overlap next V-tile copy with this block's compute
      async_copy_vtile(vbase + (size_t)(kb + 32) * LDQ, LDQ,
                       vbuf + (ibuf ^ 1) * 2048, lane);

    // S tile = Q * K^T : 16 queries x 32 keys (two 16x16 C-frags)
    v8f s[2];
#pragma unroll
    for (int f = 0; f < 2; ++f)
#pragma unroll
      for (int r = 0; r < 8; ++r) s[f][r] = 0.0f;
#pragma unroll
    for (int f = 0; f < 2; ++f)
#pragma unroll
      for (int ks = 0; ks < 2; ++ks) {
        v16bf kf = frag_from_f32(
            kbase + (size_t)(kb + f * 16 + lm) * LDQ + ks * 32, hlf);
        s[f] = wmma_bf16(qf[ks], kf, s[f]);
      }

    // scale + causal mask
#pragma unroll
    for (int f = 0; f < 2; ++f)
#pragma unroll
      for (int r = 0; r < 8; ++r) {
        float v = s[f][r] * 0.125f;  // 1/sqrt(64)
        if (mask) {
          const int kg = kb + f * 16 + lm;
          const int qg = q0 + r + 8 * hlf;
          if (kg > qg) v = -1e30f;
        }
        s[f][r] = v;
      }

    // online softmax per row (row = r + 8*half lives in one lane-half)
#pragma unroll
    for (int r = 0; r < 8; ++r) {
      float tm = fmaxf(s[0][r], s[1][r]);
#pragma unroll
      for (int x = 1; x < 16; x <<= 1) tm = fmaxf(tm, __shfl_xor(tm, x, 32));
      const float mnew = fmaxf(mrow[r], tm);
      const float corr = __expf(mrow[r] - mnew);
      mrow[r] = mnew;
      const float p0 = __expf(s[0][r] - mnew);
      const float p1 = __expf(s[1][r] - mnew);
      float ps = p0 + p1;
#pragma unroll
      for (int x = 1; x < 16; x <<= 1) ps += __shfl_xor(ps, x, 32);
      lrow[r] = lrow[r] * corr + ps;
#pragma unroll
      for (int nt = 0; nt < 4; ++nt) o[nt][r] *= corr;
      const int row = r + 8 * hlf;
      pbuf[row * 32 + lm]      = (bf16_t)p0;
      pbuf[row * 32 + 16 + lm] = (bf16_t)p1;
    }
    // cross-lane LDS dependency within the wave: drain DS before reload
    asm volatile("s_wait_dscnt 0" ::: "memory");

    v16bf pf = frag_from_lds(pbuf + lm * 32, hlf);

    // Drain this block's V tile. Async loads complete in order, so with the
    // next tile's 16 copies in flight, <=16 outstanding means ours is done.
    if (have_next)
      asm volatile("s_wait_asynccnt 16" ::: "memory");
    else
      asm volatile("s_wait_asynccnt 0" ::: "memory");

    // O += P(16x32) * V(32x64): 4 n-tiles from the staged LDS V tile
#pragma unroll
    for (int nt = 0; nt < 4; ++nt) {
      v16bf vf = frag_from_lds_f32_strided(vcur + nt * 16 + lm, 64, hlf);
      o[nt] = wmma_bf16(pf, vf, o[nt]);
    }
  }

  // normalize and write [b*S + q, h*64 + d]
#pragma unroll
  for (int r = 0; r < 8; ++r) {
    const float inv = 1.0f / lrow[r];
    const int gm = b * S + q0 + r + 8 * hlf;
#pragma unroll
    for (int nt = 0; nt < 4; ++nt)
      score[(size_t)gm * DEMB + h * DH + nt * 16 + lm] = o[nt][r] * inv;
  }
}

extern "C" void kernel_launch(void* const* d_in, const int* in_sizes, int n_in,
                              void* d_out, int out_size, void* d_ws, size_t ws_size,
                              hipStream_t stream) {
  const float* x      = (const float*)d_in[0];  // [2,2048,1024]
  const float* w_in   = (const float*)d_in[1];  // [3072,1024]
  const float* b_in   = (const float*)d_in[2];  // [3072]
  const float* w_out  = (const float*)d_in[3];  // [1024,1024]
  const float* b_out  = (const float*)d_in[4];  // [1024]
  const int*   amask  = (const int*)d_in[5];
  float* out = (float*)d_out;

  float* qkv   = (float*)d_ws;                       // 4096*3072 f32 = 48 MB
  float* score = qkv + (size_t)4096 * 3072;          // 4096*1024 f32 = 16 MB

  dim3 blk(256);
  // QKV projection: M=4096, N=3072, K=1024
  gemm_nt_bias<<<dim3(3072 / 128, 4096 / 128), blk, 0, stream>>>(
      x, w_in, b_in, qkv, 4096, 3072, 1024);
  // Attention: 4096 wave-tiles / 8 waves per block; 136KB dynamic LDS
  attn_kernel<<<dim3(512), blk, 139264, stream>>>(qkv, score, amask);
  // Output projection: M=4096, N=1024, K=1024
  gemm_nt_bias<<<dim3(1024 / 128, 4096 / 128), blk, 0, stream>>>(
      score, w_out, b_out, out, 4096, 1024, 1024);
}
